// GraphLearner_49134425866398
// MI455X (gfx1250) — matrix operand order repaired and checked by reading, not compiled, not run on hardware
//
#include <hip/hip_runtime.h>
#include <hip/hip_bf16.h>
#include <stdint.h>

#define T_N   16384
#define C_N   256
#define H_N   32
#define K_TOP 16

typedef __attribute__((ext_vector_type(16))) __bf16 v16bf;
typedef __attribute__((ext_vector_type(8)))  __bf16 v8bf;
typedef __attribute__((ext_vector_type(8)))  float  v8f;
typedef __attribute__((ext_vector_type(4)))  float  v4f;

__device__ __forceinline__ unsigned short f32_bf16(float f) {
  unsigned u = __float_as_uint(f);
  u += 0x7FFFu + ((u >> 16) & 1u);          // round-to-nearest-even
  return (unsigned short)(u >> 16);
}

__device__ __forceinline__ v16bf cat16(v8bf lo, v8bf hi) {
  return __builtin_shufflevector(lo, hi, 0,1,2,3,4,5,6,7,8,9,10,11,12,13,14,15);
}

// Fully-unrolled, statically-indexed insert into ascending top-16 list.
// Keeps vals/idxs entirely in VGPRs (no scratch). In-place ascending walk is
// safe: iteration i writes slot i, reads slot i+1 before iteration i+1 writes it.
__device__ __forceinline__ void ins_reg(float v, int ix, float (&vals)[16], int (&idxs)[16]) {
#pragma unroll
  for (int i = 0; i < 15; ++i) {
    bool up   = v > vals[i + 1];             // v belongs above slot i -> shift down
    bool here = v > vals[i];
    float nv = up ? vals[i + 1] : (here ? v : vals[i]);
    int   ni = up ? idxs[i + 1] : (here ? ix : idxs[i]);
    vals[i] = nv; idxs[i] = ni;
  }
  bool here15 = v > vals[15];
  vals[15] = here15 ? v : vals[15];
  idxs[15] = here15 ? ix : idxs[15];
}

// ---------------- K0: W (f32) -> Wbf (bf16) ----------------
__global__ void k_convW(const float* __restrict__ W, unsigned short* __restrict__ Wbf) {
  int i = blockIdx.x * 256 + threadIdx.x;
  if (i < H_N * C_N) Wbf[i] = f32_bf16(W[i]);
}

// ---------------- K1: h = x @ W^T + b, stored as bf16 ----------------
__global__ __launch_bounds__(32) void k_proj(const float* __restrict__ x,
                                             const unsigned short* __restrict__ Wbf,
                                             const float* __restrict__ b,
                                             unsigned short* __restrict__ hbf) {
  int row0 = blockIdx.x * 16;
  int col0 = blockIdx.y * 16;
  int lane = threadIdx.x & 31;
  int idx15 = lane & 15;
  int half  = (lane >> 4) & 1;
  int ka = half ? 8 : 0;
  int kb = half ? 16 : 0;
  const float* xrow = x + (size_t)(row0 + idx15) * C_N;
  const unsigned short* wrow = Wbf + (size_t)(col0 + idx15) * C_N;

  v8f c = {};
  for (int kk = 0; kk < 8; ++kk) {
    union { v16bf v; unsigned short u[16]; } A;
    v4f x0 = *(const v4f*)(xrow + kk * 32 + ka);
    v4f x1 = *(const v4f*)(xrow + kk * 32 + ka + 4);
    v4f x2 = *(const v4f*)(xrow + kk * 32 + ka + 16);
    v4f x3 = *(const v4f*)(xrow + kk * 32 + ka + 20);
#pragma unroll
    for (int i = 0; i < 4; ++i) {
      A.u[i]      = f32_bf16(x0[i]);
      A.u[4 + i]  = f32_bf16(x1[i]);
      A.u[8 + i]  = f32_bf16(x2[i]);
      A.u[12 + i] = f32_bf16(x3[i]);
    }
    v8bf blo = *(const v8bf*)(wrow + kk * 32 + ka);
    v8bf bhi = *(const v8bf*)(wrow + kk * 32 + ka + 16);
    v16bf B = cat16(blo, bhi);
    c = __builtin_amdgcn_wmma_f32_16x16x32_bf16(false, A.v, false, B, (short)0, c, false, false);
  }
  float bias = b[col0 + idx15];
  int rbase = row0 + half * 8;
#pragma unroll
  for (int r = 0; r < 8; ++r)
    hbf[(size_t)(rbase + r) * H_N + col0 + idx15] = f32_bf16(c[r] + bias);
}

// ---------------- K2: row stats, two-phase (max pass, then exp-sum pass) ----------------
__global__ __launch_bounds__(256) void k_stats(const unsigned short* __restrict__ hbf,
                                               float* __restrict__ rowmax,
                                               float* __restrict__ rowsum) {
  int row0 = blockIdx.x * 16;
  int tid = threadIdx.x;
  int lane = tid & 31, wave = tid >> 5;
  int idx15 = lane & 15;
  int half = (lane >> 4) & 1;
  int ka = half ? 8 : 0;
  int kb = half ? 16 : 0;

  const unsigned short* arow = hbf + (size_t)(row0 + idx15) * H_N;
  v16bf A = cat16(*(const v8bf*)(arow + kb), *(const v8bf*)(arow + kb + 8));

  __shared__ float red[8][16];
  __shared__ float fin[16];

  // ---- pass 1: row max ----
  float m[8];
#pragma unroll
  for (int r = 0; r < 8; ++r) m[r] = -1e30f;
  for (int j = wave; j < T_N / 16; j += 8) {
    const unsigned short* brow = hbf + (size_t)(j * 16 + idx15) * H_N;
    v16bf B = cat16(*(const v8bf*)(brow + kb), *(const v8bf*)(brow + kb + 8));
    v8f c = {};
    c = __builtin_amdgcn_wmma_f32_16x16x32_bf16(false, A, false, B, (short)0, c, false, false);
#pragma unroll
    for (int r = 0; r < 8; ++r) m[r] = fmaxf(m[r], c[r]);
  }
#pragma unroll
  for (int off = 1; off < 16; off <<= 1)
#pragma unroll
    for (int r = 0; r < 8; ++r) m[r] = fmaxf(m[r], __shfl_xor(m[r], off, 32));
  if (idx15 == 0)
#pragma unroll
    for (int r = 0; r < 8; ++r) red[wave][half * 8 + r] = m[r];
  __syncthreads();
  if (tid < 16) {
    float mm = -1e30f;
    for (int w = 0; w < 8; ++w) mm = fmaxf(mm, red[w][tid]);
    rowmax[row0 + tid] = mm;
    fin[tid] = mm;
  }
  __syncthreads();
  float mrow[8];
#pragma unroll
  for (int r = 0; r < 8; ++r) mrow[r] = fin[half * 8 + r];

  // ---- pass 2: sum of exp(v - max) ----
  float s[8];
#pragma unroll
  for (int r = 0; r < 8; ++r) s[r] = 0.f;
  for (int j = wave; j < T_N / 16; j += 8) {
    const unsigned short* brow = hbf + (size_t)(j * 16 + idx15) * H_N;
    v16bf B = cat16(*(const v8bf*)(brow + kb), *(const v8bf*)(brow + kb + 8));
    v8f c = {};
    c = __builtin_amdgcn_wmma_f32_16x16x32_bf16(false, A, false, B, (short)0, c, false, false);
#pragma unroll
    for (int r = 0; r < 8; ++r) s[r] += __expf(c[r] - mrow[r]);
  }
#pragma unroll
  for (int off = 1; off < 16; off <<= 1)
#pragma unroll
    for (int r = 0; r < 8; ++r) s[r] += __shfl_xor(s[r], off, 32);
  if (idx15 == 0)
#pragma unroll
    for (int r = 0; r < 8; ++r) red[wave][half * 8 + r] = s[r];
  __syncthreads();
  if (tid < 16) {
    float ss = 0.f;
    for (int w = 0; w < 8; ++w) ss += red[w][tid];
    rowsum[row0 + tid] = ss;
  }
}

// ---------------- K3: adj write (NT b128) + fused per-row top-16 -> edge_index ----------------
__global__ __launch_bounds__(256) void k_adj_topk(const unsigned short* __restrict__ hbf,
                                                  const float* __restrict__ rowmax,
                                                  const float* __restrict__ rowsum,
                                                  float* __restrict__ adj,
                                                  long long* __restrict__ ei) {
  int row0 = blockIdx.x * 16;
  int tid = threadIdx.x;
  int lane = tid & 31, wave = tid >> 5;
  int idx15 = lane & 15;
  int half = (lane >> 4) & 1;
  int ka = half ? 8 : 0;
  int kb = half ? 16 : 0;

  __shared__ float sT[8][16][20];        // per-wave 16x16 tile staging (padded rows)
  __shared__ float Lv[8][32][16];        // per-(wave,lane) top-16 values
  __shared__ int   Li[8][32][16];        // per-(wave,lane) top-16 indices

  const unsigned short* arow = hbf + (size_t)(row0 + idx15) * H_N;
  v16bf A = cat16(*(const v8bf*)(arow + kb), *(const v8bf*)(arow + kb + 8));

  float rm[8], ri[8];
#pragma unroll
  for (int r = 0; r < 8; ++r) {
    int row = row0 + half * 8 + r;
    rm[r] = rowmax[row];
    ri[r] = 1.0f / rowsum[row];
  }

  float tv[16]; int tix[16];
#pragma unroll
  for (int i = 0; i < 16; ++i) { tv[i] = -1e30f; tix[i] = 0; }
  float minv = -1e30f;                    // register copy of tv[0]

  for (int j = wave; j < T_N / 16; j += 8) {
    int col0 = j * 16;
    const unsigned short* brow = hbf + (size_t)(col0 + idx15) * H_N;
    v16bf B = cat16(*(const v8bf*)(brow + kb), *(const v8bf*)(brow + kb + 8));
    v8f c = {};
    c = __builtin_amdgcn_wmma_f32_16x16x32_bf16(false, A, false, B, (short)0, c, false, false);

    // stage normalized probabilities (fragment layout -> row-major)
#pragma unroll
    for (int r = 0; r < 8; ++r)
      sT[wave][half * 8 + r][idx15] = __expf(c[r] - rm[r]) * ri[r];
    asm volatile("" ::: "memory");       // keep LDS store->load order (HW LDS is in-order per wave)

    // transposed view: this lane owns row idx15, columns [half*8, half*8+8)
    v4f p0 = *(const v4f*)&sT[wave][idx15][half * 8];
    v4f p1 = *(const v4f*)&sT[wave][idx15][half * 8 + 4];

    float* dst = adj + (size_t)(row0 + idx15) * T_N + col0 + half * 8;
    __builtin_nontemporal_store(p0, (v4f*)dst);
    __builtin_nontemporal_store(p1, (v4f*)(dst + 4));

    // fast-path reject is one register compare; accepts are rare (~66/lane total)
#pragma unroll
    for (int i = 0; i < 4; ++i)
      if (p0[i] > minv) { ins_reg(p0[i], col0 + half * 8 + i, tv, tix); minv = tv[0]; }
#pragma unroll
    for (int i = 0; i < 4; ++i)
      if (p1[i] > minv) { ins_reg(p1[i], col0 + half * 8 + 4 + i, tv, tix); minv = tv[0]; }
  }

  // publish per-(wave,lane) candidate lists
#pragma unroll
  for (int i = 0; i < 16; ++i) { Lv[wave][lane][i] = tv[i]; Li[wave][lane][i] = tix[i]; }
  __syncthreads();

  // final merge: thread t owns row row0+t; merges 16 lists (8 waves x 2 halves)
  if (tid < 16) {
    float v2[16]; int i2[16];
#pragma unroll
    for (int i = 0; i < 16; ++i) { v2[i] = -1e30f; i2[i] = 0; }
    for (int w = 0; w < 8; ++w)
      for (int hf = 0; hf < 2; ++hf) {
        int ln = hf * 16 + tid;
        for (int i = 0; i < 16; ++i) ins_reg(Lv[w][ln][i], Li[w][ln][i], v2, i2);
      }
    long long* esrc = ei;
    long long* edst = ei + (long long)T_N * K_TOP;
    int row = row0 + tid;
    for (int jj = 0; jj < K_TOP; ++jj) {
      esrc[(size_t)row * K_TOP + jj] = row;
      edst[(size_t)row * K_TOP + jj] = i2[15 - jj];   // descending like lax.top_k
    }
  }
}

extern "C" void kernel_launch(void* const* d_in, const int* in_sizes, int n_in,
                              void* d_out, int out_size, void* d_ws, size_t ws_size,
                              hipStream_t stream) {
  const float* x = (const float*)d_in[0];
  const float* W = (const float*)d_in[1];
  const float* b = (const float*)d_in[2];
  // d_in[3] = top_k (known constant 16)

  float* adj = (float*)d_out;
  long long* ei = (long long*)(adj + (size_t)T_N * T_N);

  char* ws = (char*)d_ws;
  unsigned short* Wbf = (unsigned short*)ws;                                  // 16 KB
  unsigned short* hbf = (unsigned short*)(ws + 16 * 1024);                    // 1 MB
  float* rowmax = (float*)(ws + 16 * 1024 + (size_t)T_N * H_N * 2);
  float* rowsum = rowmax + T_N;

  k_convW<<<(H_N * C_N + 255) / 256, 256, 0, stream>>>(W, Wbf);
  k_proj<<<dim3(T_N / 16, H_N / 16), 32, 0, stream>>>(x, Wbf, b, hbf);
  k_stats<<<T_N / 16, 256, 0, stream>>>(hbf, rowmax, rowsum);
  k_adj_topk<<<T_N / 16, 256, 0, stream>>>(hbf, rowmax, rowsum, adj, ei);
}
